// TransformerDecoderLayer_60172491816991
// MI455X (gfx1250) — matrix operand order repaired and checked by reading, not compiled, not run on hardware
//
#include <hip/hip_runtime.h>
#include <hip/hip_bf16.h>

// ---------------------------------------------------------------------------
// Transformer decoder layer for MI455X (gfx1250, wave32, WMMA + async DMA).
// B=4, S=SE=1024, D=1024, H=16, DK=DV=64, DH=4096.
// GEMMs + attention on v_wmma_f32_16x16x32_bf16; LDS staging via
// global_load_async_to_lds_b128 with double buffering (ASYNCcnt pipeline).
// ---------------------------------------------------------------------------

#define kB   4
#define kS   1024
#define kSE  1024
#define kD   1024
#define kH   16
#define kDH  4096
#define kM   (kB * kS)
#define LN_EPS 1e-5f

typedef __attribute__((ext_vector_type(16))) __bf16 bf16x16;
typedef __attribute__((ext_vector_type(8)))  float  f32x8;

union FragU { bf16x16 v; uint4 u[2]; };

// A-fragment (16xK row-major [M][ldk]): lane m=l&15, h=l>>4;
// elems 0..7 -> K=8h+e, elems 8..15 -> K=16+8h+(e-8). Two 16B runs.
__device__ __forceinline__ bf16x16 load_a_frag(const __bf16* base, int ldk) {
  const int lane = threadIdx.x & 31;
  const int m = lane & 15, h = lane >> 4;
  const __bf16* p = base + (size_t)m * ldk + 8 * h;
  FragU f;
  f.u[0] = *(const uint4*)(p);
  f.u[1] = *(const uint4*)(p + 16);
  return f.v;
}

// B-fragment from TRANSPOSED tile [N][ldk]: lane n=l&15, h=l>>4;
// elems 0..7 -> K=8h+e, elems 8..15 -> K=16+8h+(e-8). Two 16B runs.
__device__ __forceinline__ bf16x16 load_b_fragT(const __bf16* base, int ldk) {
  const int lane = threadIdx.x & 31;
  const int n = lane & 15, h = lane >> 4;
  const __bf16* p = base + (size_t)n * ldk + 8 * h;
  FragU f;
  f.u[0] = *(const uint4*)(p);
  f.u[1] = *(const uint4*)(p + 16);
  return f.v;
}

__device__ __forceinline__ f32x8 wmma_bf16(bf16x16 a, bf16x16 b, f32x8 c) {
  return __builtin_amdgcn_wmma_f32_16x16x32_bf16(false, a, false, b, (short)0, c,
                                                 false, false);
}

// 16B global -> LDS async copy (per active lane), tracked by ASYNCcnt.
__device__ __forceinline__ void async_copy16(void* lds_ptr, const void* gptr) {
  unsigned lds_off = (unsigned)(size_t)lds_ptr;       // low 32b = LDS offset
  unsigned long long ga = (unsigned long long)gptr;
  asm volatile("global_load_async_to_lds_b128 %0, %1, off"
               :: "v"(lds_off), "v"(ga) : "memory");
}
#define ASYNC_WAIT(n) asm volatile("s_wait_asynccnt " #n ::: "memory")

// ---------------------------------------------------------------------------
// Elementwise / repack kernels (one-time per call, bandwidth-trivial)
// ---------------------------------------------------------------------------
__global__ __launch_bounds__(256) void cvt_f32_bf16_k(const float* __restrict__ in,
                                                      __bf16* __restrict__ out, int n) {
  int i = blockIdx.x * 256 + threadIdx.x;
  if (i < n) out[i] = (__bf16)in[i];
}

// Wq/Wk/Wv (H,D,64) f32 -> transposed bf16 [N=H*64][K=D]
__global__ __launch_bounds__(256) void repack_hw_t_k(const float* __restrict__ in,
                                                     __bf16* __restrict__ out) {
  int i = blockIdx.x * 256 + threadIdx.x;          // over (H*64)*D = 1M
  if (i >= kD * kH * 64) return;
  int n = i >> 10, d = i & 1023;
  int h = n >> 6, e = n & 63;
  out[i] = (__bf16)in[((size_t)h * kD + d) * 64 + e];
}

// Generic transpose: in f32 [K][N] -> out bf16 [N][K]
__global__ __launch_bounds__(256) void transpose_w_k(const float* __restrict__ in,
                                                     __bf16* __restrict__ out,
                                                     int K, int N) {
  int i = blockIdx.x * 256 + threadIdx.x;
  if (i >= K * N) return;
  int n = i / K, k = i - n * K;
  out[i] = (__bf16)in[(size_t)k * N + n];
}

// V (B,S,H*64) bf16 -> Vt [b][h][dv][s] bf16 (scatter-free flash staging)
__global__ __launch_bounds__(256) void transpose_v_k(const __bf16* __restrict__ in,
                                                     __bf16* __restrict__ out,
                                                     int Skv) {
  int i = blockIdx.x * 256 + threadIdx.x;          // over B*H*64*Skv
  if (i >= kB * kH * 64 * Skv) return;
  int s = i & (Skv - 1);
  int r = i / Skv;
  int dv = r & 63; r >>= 6;
  int h = r & 15;  r >>= 4;
  int b = r;
  out[i] = in[((size_t)(b * Skv + s)) * (kH * 64) + h * 64 + dv];
}

// ---------------------------------------------------------------------------
// Tiled bf16 GEMM: C[M,N] = A[M,K] * Wt[N,K]^T (+bias)(+relu)
// BM=128, BN=128, BK=32; double-buffered LDS fed by async-to-LDS DMA.
// 8 waves; wave tile 32x64 = 2x4 WMMA per K-step.
// ---------------------------------------------------------------------------
#define BM 128
#define BN 128
#define BK 32
#define LDT (BK + 8)     // 40 elems, 80B row stride (16B multiple)

__global__ __launch_bounds__(256) void gemm_bf16_k(
    const __bf16* __restrict__ A, const __bf16* __restrict__ Bt,
    const float* __restrict__ bias, float* __restrict__ Cf,
    __bf16* __restrict__ Cb, int M, int N, int K, int relu) {
  __shared__ __bf16 As[2][BM][LDT];
  __shared__ __bf16 Bst[2][BN][LDT];   // [n][k]

  const int t = threadIdx.x;
  const int w = t >> 5, lane = t & 31;
  const int bm = blockIdx.y * BM, bn = blockIdx.x * BN;
  const int wm = (w & 3) * 32, wn = (w >> 2) * 64;
  const int nn = lane & 15, hh = lane >> 4;

  // this thread's two staging chunks (r, c) covering 128x32 in 16B units
  int r0 = t >> 2, c0 = (t & 3) * 8;                  // ids 0..255
  int r1 = (256 + t) >> 2, c1 = ((256 + t) & 3) * 8;  // ids 256..511

  f32x8 acc[2][4] = {};
  const int ntiles = K / BK;

  // prologue: tile 0 into buffer 0 (4 async ops/thread)
  async_copy16(&As[0][r0][c0],  A + (size_t)(bm + r0) * K + c0);
  async_copy16(&Bst[0][r0][c0], Bt + (size_t)(bn + r0) * K + c0);
  async_copy16(&As[0][r1][c1],  A + (size_t)(bm + r1) * K + c1);
  async_copy16(&Bst[0][r1][c1], Bt + (size_t)(bn + r1) * K + c1);

  for (int kt = 0; kt < ntiles; ++kt) {
    const int cur = kt & 1;
    if (kt + 1 < ntiles) {
      const int k1 = (kt + 1) * BK;
      const int nb = cur ^ 1;
      async_copy16(&As[nb][r0][c0],  A + (size_t)(bm + r0) * K + k1 + c0);
      async_copy16(&Bst[nb][r0][c0], Bt + (size_t)(bn + r0) * K + k1 + c0);
      async_copy16(&As[nb][r1][c1],  A + (size_t)(bm + r1) * K + k1 + c1);
      async_copy16(&Bst[nb][r1][c1], Bt + (size_t)(bn + r1) * K + k1 + c1);
      __builtin_prefetch((const void*)(A + (size_t)(bm + r0) * K + k1 + BK + c0), 0, 0);
      __builtin_prefetch((const void*)(Bt + (size_t)(bn + r0) * K + k1 + BK + c0), 0, 0);
      ASYNC_WAIT(4);           // 4 newest belong to tile kt+1 -> tile kt done
    } else {
      ASYNC_WAIT(0);
    }
    __syncthreads();

    bf16x16 af0 = load_a_frag(&As[cur][wm][0], LDT);
    bf16x16 af1 = load_a_frag(&As[cur][wm + 16][0], LDT);
#pragma unroll
    for (int j = 0; j < 4; ++j) {
      bf16x16 bf = load_b_fragT(&Bst[cur][wn + j * 16][0], LDT);
      acc[0][j] = wmma_bf16(af0, bf, acc[0][j]);
      acc[1][j] = wmma_bf16(af1, bf, acc[1][j]);
    }
    __syncthreads();           // all waves done reading buf cur
  }

#pragma unroll
  for (int i = 0; i < 2; ++i)
#pragma unroll
    for (int j = 0; j < 4; ++j) {
      int gn = bn + wn + j * 16 + nn;
      float bv = bias ? bias[gn] : 0.f;
#pragma unroll
      for (int r = 0; r < 8; ++r) {
        int gm = bm + wm + i * 16 + r + 8 * hh;
        float v = acc[i][j][r] + bv;
        if (relu) v = fmaxf(v, 0.f);
        size_t idx = (size_t)gm * N + gn;
        if (Cf) Cf[idx] = v;
        if (Cb) Cb[idx] = (__bf16)v;
      }
    }
}

// ---------------------------------------------------------------------------
// Flash attention: block = (b,h,128 q-rows); 8 waves x 16 q-rows.
// Q,K in (B,S,H*64) bf16; V pre-transposed [b][h][dv][Skv].
// K/V tiles double-buffered, fed by async-to-LDS DMA.
// ---------------------------------------------------------------------------
__global__ __launch_bounds__(256) void flash_attn_k(
    const __bf16* __restrict__ Qg, const __bf16* __restrict__ Kg,
    const __bf16* __restrict__ Vt, __bf16* __restrict__ Og,
    int Skv, int causal) {
  __shared__ __bf16 Ks[2][32][64 + 8];     // [key][dk]
  __shared__ __bf16 Vst[2][64][32 + 8];    // [dv][key]
  __shared__ __bf16 Ps[8][16][32 + 8];     // per-wave P tile [q][key]

  const int t = threadIdx.x, w = t >> 5, lane = t & 31;
  const int nn = lane & 15, hh = lane >> 4;
  const int qc = blockIdx.x & 7;
  const int h  = (blockIdx.x >> 3) & 15;
  const int b  = blockIdx.x >> 7;
  const int q0 = qc * 128 + w * 16;

  const __bf16* qbase = Qg + ((size_t)(b * kS + q0)) * (kH * 64) + h * 64;
  bf16x16 aq[2];
#pragma unroll
  for (int tk = 0; tk < 2; ++tk) aq[tk] = load_a_frag(qbase + tk * 32, kH * 64);

  const int key = t >> 3, c8 = (t & 7) * 8;     // K-tile chunk (32x64)
  const int dv  = t >> 2, kc = (t & 3) * 8;     // V-tile chunk (64x32)
  const __bf16* kgb = Kg + ((size_t)(b * Skv)) * (kH * 64) + h * 64;
  const __bf16* vgb = Vt + ((size_t)((b * kH + h) * 64 + dv)) * Skv;

  float mrow[8], lrow[8];
#pragma unroll
  for (int r = 0; r < 8; ++r) { mrow[r] = -1e30f; lrow[r] = 0.f; }
  f32x8 o[4] = {};

  const int nsteps = causal ? (qc * 4 + 4) : (Skv / 32);

  // prologue: tile 0 into buffer 0 (2 async ops/thread)
  async_copy16(&Ks[0][key][c8], kgb + (size_t)key * (kH * 64) + c8);
  async_copy16(&Vst[0][dv][kc], vgb + kc);

  for (int kb = 0; kb < nsteps; ++kb) {
    const int kbase = kb * 32;
    const int cur = kb & 1;
    if (kb + 1 < nsteps) {
      const int nb = cur ^ 1, kn = kbase + 32;
      async_copy16(&Ks[nb][key][c8], kgb + (size_t)(kn + key) * (kH * 64) + c8);
      async_copy16(&Vst[nb][dv][kc], vgb + kn + kc);
      ASYNC_WAIT(2);
    } else {
      ASYNC_WAIT(0);
    }
    __syncthreads();

    f32x8 s[2] = {};
#pragma unroll
    for (int tk = 0; tk < 2; ++tk)
#pragma unroll
      for (int j = 0; j < 2; ++j) {
        bf16x16 bk = load_b_fragT(&Ks[cur][j * 16][tk * 32], 64 + 8);
        s[j] = wmma_bf16(aq[tk], bk, s[j]);
      }

#pragma unroll
    for (int r = 0; r < 8; ++r) {
      float v0 = s[0][r] * 0.125f;
      float v1 = s[1][r] * 0.125f;
      if (causal) {
        int qrow = q0 + r + 8 * hh;
        if (kbase + nn      > qrow) v0 = -1e30f;
        if (kbase + 16 + nn > qrow) v1 = -1e30f;
      }
      float tm = fmaxf(v0, v1);
#pragma unroll
      for (int off = 8; off >= 1; off >>= 1) tm = fmaxf(tm, __shfl_xor(tm, off, 32));
      float mn = fmaxf(mrow[r], tm);
      float alpha = __expf(mrow[r] - mn);
      float p0 = __expf(v0 - mn);
      float p1 = __expf(v1 - mn);
      float ps = p0 + p1;
#pragma unroll
      for (int off = 8; off >= 1; off >>= 1) ps += __shfl_xor(ps, off, 32);
      lrow[r] = lrow[r] * alpha + ps;
      mrow[r] = mn;
#pragma unroll
      for (int v = 0; v < 4; ++v) o[v][r] *= alpha;
      Ps[w][r + 8 * hh][nn]      = (__bf16)p0;
      Ps[w][r + 8 * hh][16 + nn] = (__bf16)p1;
    }

    asm volatile("s_wait_dscnt 0" ::: "memory");  // own-wave P visibility

    bf16x16 pa = load_a_frag(&Ps[w][0][0], 32 + 8);
#pragma unroll
    for (int v = 0; v < 4; ++v) {
      bf16x16 bv = load_b_fragT(&Vst[cur][v * 16][0], 32 + 8);
      o[v] = wmma_bf16(pa, bv, o[v]);
    }
    __syncthreads();
  }

#pragma unroll
  for (int v = 0; v < 4; ++v)
#pragma unroll
    for (int r = 0; r < 8; ++r) {
      int qrow = q0 + r + 8 * hh;
      int dvi = v * 16 + nn;
      float val = o[v][r] / lrow[r];
      Og[((size_t)(b * kS + qrow)) * (kH * 64) + h * 64 + dvi] = (__bf16)val;
    }
}

// ---------------------------------------------------------------------------
// Fused residual-add + LayerNorm (one block per 1024-col row)
// ---------------------------------------------------------------------------
__global__ __launch_bounds__(256) void add_ln_k(
    const float* __restrict__ a, const float* __restrict__ res,
    const float* __restrict__ g, const float* __restrict__ be,
    float* __restrict__ outf, __bf16* __restrict__ outb) {
  __shared__ float red[8];
  const int t = threadIdx.x, w = t >> 5, lane = t & 31;
  const size_t row = blockIdx.x;
  const int c0 = t * 4;

  float4 va = *(const float4*)(a + row * kD + c0);
  float4 vr = *(const float4*)(res + row * kD + c0);
  float x[4] = {va.x + vr.x, va.y + vr.y, va.z + vr.z, va.w + vr.w};

  float s = x[0] + x[1] + x[2] + x[3];
#pragma unroll
  for (int off = 16; off >= 1; off >>= 1) s += __shfl_xor(s, off, 32);
  if (lane == 0) red[w] = s;
  __syncthreads();
  float tot = 0.f;
#pragma unroll
  for (int i = 0; i < 8; ++i) tot += red[i];
  float mean = tot * (1.f / kD);
  __syncthreads();

  float vs = 0.f;
#pragma unroll
  for (int i = 0; i < 4; ++i) { float d = x[i] - mean; vs += d * d; }
#pragma unroll
  for (int off = 16; off >= 1; off >>= 1) vs += __shfl_xor(vs, off, 32);
  if (lane == 0) red[w] = vs;
  __syncthreads();
  float vtot = 0.f;
#pragma unroll
  for (int i = 0; i < 8; ++i) vtot += red[i];
  float rinv = rsqrtf(vtot * (1.f / kD) + LN_EPS);

#pragma unroll
  for (int i = 0; i < 4; ++i) {
    int c = c0 + i;
    float y = g[c] * (x[i] - mean) * rinv + be[c];
    outf[row * kD + c] = y;
    if (outb) outb[row * kD + c] = (__bf16)y;
  }
}

// ---------------------------------------------------------------------------
// Host orchestration
// ---------------------------------------------------------------------------
static void launch_gemm(const __bf16* A, const __bf16* Bt, const float* bias,
                        float* Cf, __bf16* Cb, int M, int N, int K, int relu,
                        hipStream_t s) {
  dim3 g(N / BN, M / BM);
  gemm_bf16_k<<<g, 256, 0, s>>>(A, Bt, bias, Cf, Cb, M, N, K, relu);
}

extern "C" void kernel_launch(void* const* d_in, const int* in_sizes, int n_in,
                              void* d_out, int out_size, void* d_ws, size_t ws_size,
                              hipStream_t stream) {
  const float* x_in = (const float*)d_in[0];
  const float* enc  = (const float*)d_in[2];
  const float* Wq1 = (const float*)d_in[4];  const float* bq1 = (const float*)d_in[5];
  const float* Wk1 = (const float*)d_in[6];  const float* bk1 = (const float*)d_in[7];
  const float* Wv1 = (const float*)d_in[8];  const float* bv1 = (const float*)d_in[9];
  const float* Wo1 = (const float*)d_in[10]; const float* bo1 = (const float*)d_in[11];
  const float* Wq2 = (const float*)d_in[12]; const float* bq2 = (const float*)d_in[13];
  const float* Wk2 = (const float*)d_in[14]; const float* bk2 = (const float*)d_in[15];
  const float* Wv2 = (const float*)d_in[16]; const float* bv2 = (const float*)d_in[17];
  const float* Wo2 = (const float*)d_in[18]; const float* bo2 = (const float*)d_in[19];
  const float* W1f = (const float*)d_in[20]; const float* b1f = (const float*)d_in[21];
  const float* W2f = (const float*)d_in[22]; const float* b2f = (const float*)d_in[23];
  const float* g1 = (const float*)d_in[24];  const float* be1 = (const float*)d_in[25];
  const float* g2 = (const float*)d_in[26];  const float* be2 = (const float*)d_in[27];
  const float* g3 = (const float*)d_in[28];  const float* be3 = (const float*)d_in[29];

  char* ws = (char*)d_ws;
  size_t off = 0;
  auto take = [&](size_t bytes) -> char* {
    char* p = ws + off;
    off += (bytes + 255) & ~(size_t)255;
    return p;
  };
  const size_t W_DD  = (size_t)kD * kD * 2;
  const size_t W_DDH = (size_t)kD * kDH * 2;
  const size_t ACT_B = (size_t)kM * kD * 2;
  const size_t ACT_F = (size_t)kM * kD * 4;
  const size_t HID_B = (size_t)kM * kDH * 2;

  __bf16* Wq1b = (__bf16*)take(W_DD);
  __bf16* Wk1b = (__bf16*)take(W_DD);
  __bf16* Wv1b = (__bf16*)take(W_DD);
  __bf16* Wo1b = (__bf16*)take(W_DD);
  __bf16* Wq2b = (__bf16*)take(W_DD);
  __bf16* Wk2b = (__bf16*)take(W_DD);
  __bf16* Wv2b = (__bf16*)take(W_DD);
  __bf16* Wo2b = (__bf16*)take(W_DD);
  __bf16* W1fb = (__bf16*)take(W_DDH);
  __bf16* W2fb = (__bf16*)take(W_DDH);
  __bf16* xb   = (__bf16*)take(ACT_B);
  __bf16* eb   = (__bf16*)take(ACT_B);
  __bf16* Qb   = (__bf16*)take(ACT_B);
  __bf16* Kb   = (__bf16*)take(ACT_B);
  __bf16* Vb   = (__bf16*)take(ACT_B);
  __bf16* Vtb  = (__bf16*)take(ACT_B);
  __bf16* Ab   = (__bf16*)take(ACT_B);
  float*  T0   = (float*)take(ACT_F);
  float*  x1f  = (float*)take(ACT_F);
  __bf16* x1b  = (__bf16*)take(ACT_B);
  float*  x2f  = (float*)take(ACT_F);
  __bf16* x2b  = (__bf16*)take(ACT_B);
  __bf16* Hb   = (__bf16*)take(HID_B);
  (void)off; (void)ws_size; (void)in_sizes; (void)n_in; (void)out_size;

  const int nW  = kD * kD;
  const int nWH = kD * kDH;
  const int nA  = kM * kD;

  // 1) weight repack/transpose to bf16 [N][K]
  repack_hw_t_k<<<(nW + 255) / 256, 256, 0, stream>>>(Wq1, Wq1b);
  repack_hw_t_k<<<(nW + 255) / 256, 256, 0, stream>>>(Wk1, Wk1b);
  repack_hw_t_k<<<(nW + 255) / 256, 256, 0, stream>>>(Wv1, Wv1b);
  repack_hw_t_k<<<(nW + 255) / 256, 256, 0, stream>>>(Wq2, Wq2b);
  repack_hw_t_k<<<(nW + 255) / 256, 256, 0, stream>>>(Wk2, Wk2b);
  repack_hw_t_k<<<(nW + 255) / 256, 256, 0, stream>>>(Wv2, Wv2b);
  transpose_w_k<<<(nW + 255) / 256, 256, 0, stream>>>(Wo1, Wo1b, kD, kD);
  transpose_w_k<<<(nW + 255) / 256, 256, 0, stream>>>(Wo2, Wo2b, kD, kD);
  transpose_w_k<<<(nWH + 255) / 256, 256, 0, stream>>>(W1f, W1fb, kD, kDH);
  transpose_w_k<<<(nWH + 255) / 256, 256, 0, stream>>>(W2f, W2fb, kDH, kD);
  cvt_f32_bf16_k<<<(nA + 255) / 256, 256, 0, stream>>>(x_in, xb, nA);
  cvt_f32_bf16_k<<<(nA + 255) / 256, 256, 0, stream>>>(enc, eb, nA);

  // 2) masked self-attention
  launch_gemm(xb, Wq1b, bq1, nullptr, Qb, kM, kD, kD, 0, stream);
  launch_gemm(xb, Wk1b, bk1, nullptr, Kb, kM, kD, kD, 0, stream);
  launch_gemm(xb, Wv1b, bv1, nullptr, Vb, kM, kD, kD, 0, stream);
  transpose_v_k<<<(nA + 255) / 256, 256, 0, stream>>>(Vb, Vtb, kS);
  flash_attn_k<<<kB * kH * (kS / 128), 256, 0, stream>>>(Qb, Kb, Vtb, Ab, kS, 1);
  launch_gemm(Ab, Wo1b, bo1, T0, nullptr, kM, kD, kD, 0, stream);
  add_ln_k<<<kM, 256, 0, stream>>>(T0, x_in, g1, be1, x1f, x1b);

  // 3) cross-attention
  launch_gemm(x1b, Wq2b, bq2, nullptr, Qb, kM, kD, kD, 0, stream);
  launch_gemm(eb,  Wk2b, bk2, nullptr, Kb, kM, kD, kD, 0, stream);
  launch_gemm(eb,  Wv2b, bv2, nullptr, Vb, kM, kD, kD, 0, stream);
  transpose_v_k<<<(nA + 255) / 256, 256, 0, stream>>>(Vb, Vtb, kSE);
  flash_attn_k<<<kB * kH * (kS / 128), 256, 0, stream>>>(Qb, Kb, Vtb, Ab, kSE, 0);
  launch_gemm(Ab, Wo2b, bo2, T0, nullptr, kM, kD, kD, 0, stream);
  add_ln_k<<<kM, 256, 0, stream>>>(T0, x1f, g2, be2, x2f, x2b);

  // 4) FFN
  launch_gemm(x2b, W1fb, b1f, nullptr, Hb, kM, kDH, kD, 1, stream);
  launch_gemm(Hb, W2fb, b2f, T0, nullptr, kM, kD, kDH, 0, stream);
  add_ln_k<<<kM, 256, 0, stream>>>(T0, x2f, g3, be3, (float*)d_out, nullptr);
}
